// Drnet_9234179686929
// MI455X (gfx1250) — matrix operand order repaired and checked by reading, not compiled, not run on hardware
//
#include <hip/hip_runtime.h>

// ---------------- types ----------------
typedef __attribute__((ext_vector_type(16))) __bf16          v16bf;
typedef __attribute__((ext_vector_type(8)))  float           v8f;

union Frag {
    v16bf bf;
    uint4 q[2];
};

#define WMMA_BF16(A, B, C) \
    __builtin_amdgcn_wmma_f32_16x16x32_bf16(false, (A), false, (B), (short)0, (C), false, false)

// ---------------- geometry ----------------
#define NSAMP   1000000
#define NTILES  62500        // 1e6 / 16 exactly
#define NBLOCKS 1024

// LDS weight-image layout (bytes). Row strides padded (272 = 17*16, 144 = 9*16)
// so the 16 lanes of a B-fragment (stride n*rowstride) spread across LDS banks.
#define OFF_DW1T 0            // 64 rows(n) x 272B  (k: 128 slots, 100 valid)
#define OFF_DW2T 17408        // 64 x 144B
#define OFF_HW1T 26624        // 5 * 64 x 144B
#define OFF_HW2T 72704        // 5 * 64 x 144B
#define OFF_HW3T 118784       // 5 * 16 x 144B (row n=0 = w3, rest zero)
#define OFF_VEC  130304       // 1424 f32: db1,db2, per-q{hb1,htw1,hb2,htw2}, htw3[5], hb3[5]
#define IMG_BYTES 136000
#define IMG_U4    8500        // IMG_BYTES/16
#define STAGE_OFF 136000      // per-wave: hbuf 2304B + abuf 2304B
#define WAVE_STAGE 4608
#define TOTAL_LDS (STAGE_OFF + 8 * WAVE_STAGE)   // 172864 B (< 320KB WGP LDS)

// ---------------- helpers ----------------

// A-matrix 16x32 bf16 fragment (ISA 7.12.2): lanes 0-15 M=lane, K = {0..7,16..23}(+32*ks);
// lanes 16-31 M=lane-16, K = {8..15,24..31}(+32*ks). Source rows are 16B-aligned in LDS.
__device__ __forceinline__ Frag lds_a_frag(const char* base, int strideB, int ks, int lane) {
    int m  = lane & 15;
    int k0 = ks * 32 + ((lane & 16) ? 8 : 0);
    const char* p = base + m * strideB + k0 * 2;
    Frag f;
    f.q[0] = *(const uint4*)p;          // K = k0..k0+7
    f.q[1] = *(const uint4*)(p + 32);   // K = k0+16..k0+23
    return f;
}

// B-matrix 32x16 bf16 fragment (same K order as A): lanes 0-15 hold K={0..7,16..23},
// lanes 16-31 hold K={8..15,24..31} (+32*ks), column N = lane&15. Weights stored transposed [n][k].
__device__ __forceinline__ Frag lds_b_frag(const char* base, int strideB, int ks, int lane) {
    int n  = lane & 15;
    int k0 = ks * 32 + ((lane & 16) ? 8 : 0);
    const char* p = base + n * strideB + k0 * 2;
    Frag f;
    f.q[0] = *(const uint4*)p;          // K = k0..k0+7
    f.q[1] = *(const uint4*)(p + 32);   // K = k0+16..k0+23
    return f;
}

// C/D 16x16 f32 tile: VGPR r -> M = r + (lane&16 ? 8:0), N = lane&15.
// Trunk epilogue: relu(c + bias) -> bf16 staging row-major.
__device__ __forceinline__ void epi_b(char* buf, v8f c, int moff, int n, float bias) {
#pragma unroll
    for (int r = 0; r < 8; ++r) {
        float v = c[r] + bias;
        v = v > 0.f ? v : 0.f;
        *(__bf16*)(buf + (moff + r) * 144 + n * 2) = (__bf16)v;
    }
}

// Head epilogue: relu(c + bias + tw * t_row) -> bf16 staging row-major.
__device__ __forceinline__ void epi_bt(char* buf, v8f c, int moff, int n,
                                       float bias, float tw, const float tv[8]) {
#pragma unroll
    for (int r = 0; r < 8; ++r) {
        float v = c[r] + bias + tw * tv[r];
        v = v > 0.f ? v : 0.f;
        *(__bf16*)(buf + (moff + r) * 144 + n * 2) = (__bf16)v;
    }
}

// Per-wave LDS store->load ordering (HW keeps DS in order per wave; the wait +
// memory clobber also pins the compiler's scheduling).
__device__ __forceinline__ void lds_order() {
    asm volatile("s_wait_dscnt 0" ::: "memory");
}

// ---------------- weight repack: f32 global -> bf16 transposed image in workspace ----------------
#define PREP_N (8192 + 4096 + 20480 + 20480 + 5120 + 1424)   // 59792

__global__ __launch_bounds__(256) void drnet_prep(
    const float* __restrict__ dW1, const float* __restrict__ dW2,
    const float* __restrict__ hw1, const float* __restrict__ hw2,
    const float* __restrict__ hw3, const float* __restrict__ db1,
    const float* __restrict__ db2, const float* __restrict__ hb1,
    const float* __restrict__ htw1, const float* __restrict__ hb2,
    const float* __restrict__ htw2, const float* __restrict__ htw3,
    const float* __restrict__ hb3, char* __restrict__ img)
{
    int e = blockIdx.x * blockDim.x + threadIdx.x;
    if (e < 8192) {                                   // dW1^T, K zero-padded to 128
        int n = e >> 7, k = e & 127;
        float v = (k < 100) ? dW1[k * 64 + n] : 0.f;
        *(__bf16*)(img + OFF_DW1T + n * 272 + k * 2) = (__bf16)v;
        return;
    }
    e -= 8192;
    if (e < 4096) {                                   // dW2^T
        int n = e >> 6, k = e & 63;
        *(__bf16*)(img + OFF_DW2T + n * 144 + k * 2) = (__bf16)dW2[k * 64 + n];
        return;
    }
    e -= 4096;
    if (e < 20480) {                                  // hw1^T per head
        int q = e >> 12, n = (e >> 6) & 63, k = e & 63;
        *(__bf16*)(img + OFF_HW1T + q * 9216 + n * 144 + k * 2) =
            (__bf16)hw1[q * 4096 + k * 64 + n];
        return;
    }
    e -= 20480;
    if (e < 20480) {                                  // hw2^T per head
        int q = e >> 12, n = (e >> 6) & 63, k = e & 63;
        *(__bf16*)(img + OFF_HW2T + q * 9216 + n * 144 + k * 2) =
            (__bf16)hw2[q * 4096 + k * 64 + n];
        return;
    }
    e -= 20480;
    if (e < 5120) {                                   // hw3 as 64x16 B tile, col0 only
        int q = e >> 10, n = (e >> 6) & 15, k = e & 63;
        float v = (n == 0) ? hw3[q * 64 + k] : 0.f;
        *(__bf16*)(img + OFF_HW3T + q * 2304 + n * 144 + k * 2) = (__bf16)v;
        return;
    }
    e -= 5120;
    if (e < 1424) {                                   // f32 bias / time-weight vectors
        float v = 0.f;
        if (e < 64)        v = db1[e];
        else if (e < 128)  v = db2[e - 64];
        else if (e < 1408) {
            int j = e - 128, q = j >> 8, r = j & 255, sel = r >> 6, i = r & 63;
            v = (sel == 0) ? hb1[q * 64 + i] : (sel == 1) ? htw1[q * 64 + i]
              : (sel == 2) ? hb2[q * 64 + i] : htw2[q * 64 + i];
        }
        else if (e < 1413) v = htw3[e - 1408];
        else if (e < 1418) v = hb3[e - 1413];
        *(float*)(img + OFF_VEC + e * 4) = v;
    }
}

// ---------------- fused main kernel ----------------
__global__ __launch_bounds__(256) void drnet_main(
    const float* __restrict__ t, const float* __restrict__ x,
    const char* __restrict__ img, float* __restrict__ out)
{
    extern __shared__ char smem[];
    const int tid = threadIdx.x;
    for (int i = tid; i < IMG_U4; i += 256)
        ((uint4*)smem)[i] = ((const uint4*)img)[i];
    __syncthreads();

    const int wave = tid >> 5;
    const int lane = tid & 31;
    const int moff = (lane & 16) ? 8 : 0;
    const int nl   = lane & 15;
    char* hbuf = smem + STAGE_OFF + wave * WAVE_STAGE;   // trunk output (16x64 bf16)
    char* abuf = hbuf + 2304;                            // head scratch  (16x64 bf16)
    const float* vec = (const float*)(smem + OFF_VEC);

    for (int tile = blockIdx.x * 8 + wave; tile < NTILES; tile += NBLOCKS * 8) {
        const int row0 = tile * 16;

        // per-row time value + head bin (kept in VGPRs, reused by every epilogue)
        float tv[8]; int bins[8];
#pragma unroll
        for (int r = 0; r < 8; ++r) {
            float tr = t[row0 + moff + r];
            tv[r] = tr;
            int b = (int)(tr * 5.0f);
            bins[r] = b < 0 ? 0 : (b > 4 ? 4 : b);
        }

        int nxt = tile + NBLOCKS * 8;
        if (nxt < NTILES)
            __builtin_prefetch(x + (size_t)(nxt * 16 + nl) * 100, 0, 0);

        // ---- A fragments of x (f32 global -> bf16 via native cvt), K padded 100 -> 128
        const float* xr = x + (size_t)(row0 + nl) * 100;
        Frag ax[4];
#pragma unroll
        for (int j = 0; j < 4; ++j) {
            int k0 = j * 32 + ((lane & 16) ? 8 : 0);
            float f[16];
            if (j < 3) {                       // fully in range: k0+23 <= 95 < 100
                float4 a0 = *(const float4*)(xr + k0);
                float4 a1 = *(const float4*)(xr + k0 + 4);
                float4 a2 = *(const float4*)(xr + k0 + 16);
                float4 a3 = *(const float4*)(xr + k0 + 20);
                f[0] = a0.x;  f[1] = a0.y;  f[2] = a0.z;  f[3] = a0.w;
                f[4] = a1.x;  f[5] = a1.y;  f[6] = a1.z;  f[7] = a1.w;
                f[8] = a2.x;  f[9] = a2.y;  f[10] = a2.z; f[11] = a2.w;
                f[12] = a3.x; f[13] = a3.y; f[14] = a3.z; f[15] = a3.w;
            } else {                           // guarded tail (k >= 100 -> 0)
#pragma unroll
                for (int i = 0; i < 8; ++i) { int k = k0 + i;      f[i]     = (k < 100) ? xr[k] : 0.f; }
#pragma unroll
                for (int i = 0; i < 8; ++i) { int k = k0 + 16 + i; f[8 + i] = (k < 100) ? xr[k] : 0.f; }
            }
#pragma unroll
            for (int i = 0; i < 16; ++i) ax[j].bf[i] = (__bf16)f[i];   // -> v_cvt_(pk_)bf16_f32
        }

        // ---- layer 1: relu(x @ dW1 + db1) -> abuf
#pragma unroll
        for (int nt = 0; nt < 4; ++nt) {
            v8f c = {};
#pragma unroll
            for (int j = 0; j < 4; ++j) {
                Frag b = lds_b_frag(smem + OFF_DW1T + nt * (16 * 272), 272, j, lane);
                c = WMMA_BF16(ax[j].bf, b.bf, c);
            }
            int n = nt * 16 + nl;
            epi_b(abuf, c, moff, n, vec[n]);
        }
        lds_order();

        // ---- layer 2: relu(h @ dW2 + db2) -> hbuf
        Frag a0 = lds_a_frag(abuf, 144, 0, lane);
        Frag a1 = lds_a_frag(abuf, 144, 1, lane);
#pragma unroll
        for (int nt = 0; nt < 4; ++nt) {
            v8f c = {};
            Frag b0 = lds_b_frag(smem + OFF_DW2T + nt * (16 * 144), 144, 0, lane);
            c = WMMA_BF16(a0.bf, b0.bf, c);
            Frag b1 = lds_b_frag(smem + OFF_DW2T + nt * (16 * 144), 144, 1, lane);
            c = WMMA_BF16(a1.bf, b1.bf, c);
            int n = nt * 16 + nl;
            epi_b(hbuf, c, moff, n, vec[64 + n]);
        }
        lds_order();

        Frag h0 = lds_a_frag(hbuf, 144, 0, lane);
        Frag h1 = lds_a_frag(hbuf, 144, 1, lane);

        // ---- 5 heads; one of them owns each row (bin select at the end)
#pragma unroll 1
        for (int q = 0; q < 5; ++q) {
            const char* w1 = smem + OFF_HW1T + q * 9216;
            const char* w2 = smem + OFF_HW2T + q * 9216;
            const char* w3 = smem + OFF_HW3T + q * 2304;
            const float* hv = vec + 128 + q * 256;   // hb1 | htw1 | hb2 | htw2

            // a1 = relu(h @ hw1 + t*htw1 + hb1) -> abuf
#pragma unroll
            for (int nt = 0; nt < 4; ++nt) {
                v8f c = {};
                Frag b0 = lds_b_frag(w1 + nt * (16 * 144), 144, 0, lane);
                c = WMMA_BF16(h0.bf, b0.bf, c);
                Frag b1 = lds_b_frag(w1 + nt * (16 * 144), 144, 1, lane);
                c = WMMA_BF16(h1.bf, b1.bf, c);
                int n = nt * 16 + nl;
                epi_bt(abuf, c, moff, n, hv[n], hv[64 + n], tv);
            }
            lds_order();
            Frag s0 = lds_a_frag(abuf, 144, 0, lane);
            Frag s1 = lds_a_frag(abuf, 144, 1, lane);

            // a2 = relu(a1 @ hw2 + t*htw2 + hb2) -> abuf (a1 frags already in regs)
#pragma unroll
            for (int nt = 0; nt < 4; ++nt) {
                v8f c = {};
                Frag b0 = lds_b_frag(w2 + nt * (16 * 144), 144, 0, lane);
                c = WMMA_BF16(s0.bf, b0.bf, c);
                Frag b1 = lds_b_frag(w2 + nt * (16 * 144), 144, 1, lane);
                c = WMMA_BF16(s1.bf, b1.bf, c);
                int n = nt * 16 + nl;
                epi_bt(abuf, c, moff, n, hv[128 + n], hv[192 + n], tv);
            }
            lds_order();
            Frag z0 = lds_a_frag(abuf, 144, 0, lane);
            Frag z1 = lds_a_frag(abuf, 144, 1, lane);

            // a3 = a2 @ hw3 (w3 embedded in column 0 of a B tile)
            v8f c3 = {};
            Frag b30 = lds_b_frag(w3, 144, 0, lane);
            c3 = WMMA_BF16(z0.bf, b30.bf, c3);
            Frag b31 = lds_b_frag(w3, 144, 1, lane);
            c3 = WMMA_BF16(z1.bf, b31.bf, c3);

            float w3t = vec[1408 + q], b3s = vec[1413 + q];
            if (nl == 0) {                      // column 0 lives in lanes 0 and 16
#pragma unroll
                for (int r = 0; r < 8; ++r)
                    if (bins[r] == q)
                        out[row0 + moff + r] = c3[r] + tv[r] * w3t + b3s;
            }
        }
    }
}

// ---------------- host launcher ----------------
extern "C" void kernel_launch(void* const* d_in, const int* in_sizes, int n_in,
                              void* d_out, int out_size, void* d_ws, size_t ws_size,
                              hipStream_t stream) {
    (void)in_sizes; (void)n_in; (void)out_size; (void)ws_size;
    const float* t    = (const float*)d_in[0];
    const float* x    = (const float*)d_in[1];
    const float* dW1  = (const float*)d_in[2];
    const float* db1  = (const float*)d_in[3];
    const float* dW2  = (const float*)d_in[4];
    const float* db2  = (const float*)d_in[5];
    const float* hw1  = (const float*)d_in[6];
    const float* htw1 = (const float*)d_in[7];
    const float* hb1  = (const float*)d_in[8];
    const float* hw2  = (const float*)d_in[9];
    const float* htw2 = (const float*)d_in[10];
    const float* hb2  = (const float*)d_in[11];
    const float* hw3  = (const float*)d_in[12];
    const float* htw3 = (const float*)d_in[13];
    const float* hb3  = (const float*)d_in[14];
    char* img = (char*)d_ws;

    drnet_prep<<<(PREP_N + 255) / 256, 256, 0, stream>>>(
        dW1, dW2, hw1, hw2, hw3, db1, db2, hb1, htw1, hb2, htw2, htw3, hb3, img);

    drnet_main<<<NBLOCKS, 256, TOTAL_LDS, stream>>>(t, x, img, (float*)d_out);
}
